// ResNet_23648089932289
// MI455X (gfx1250) — matrix operand order repaired and checked by reading, not compiled, not run on hardware
//
#include <hip/hip_runtime.h>
#include <hip/hip_bf16.h>
#include <cstddef>

// ---------- types ----------
typedef __attribute__((ext_vector_type(16))) __bf16 v16bf;
typedef __attribute__((ext_vector_type(8)))  __bf16 v8bf;
typedef __attribute__((ext_vector_type(8)))  float  v8f;

__device__ __forceinline__ unsigned short f2bf(float f) {
    union { float f; unsigned u; } c; c.f = f;
    unsigned u = c.u;
    u += 0x7fffu + ((u >> 16) & 1u);   // round-to-nearest-even
    return (unsigned short)(u >> 16);
}
__device__ __forceinline__ float bf2f(unsigned short h) {
    union { unsigned u; float f; } c; c.u = ((unsigned)h) << 16;
    return c.f;
}

// ---------- BN fold: s = g*rsqrt(v+eps), t = b - m*s ----------
__global__ void bnfold_kernel(const float* __restrict__ g, const float* __restrict__ b,
                              const float* __restrict__ m, const float* __restrict__ v,
                              float* __restrict__ s, float* __restrict__ t, int c) {
    int i = blockIdx.x * blockDim.x + threadIdx.x;
    if (i < c) {
        float sc = g[i] * rsqrtf(v[i] + 1e-5f);
        s[i] = sc;
        t[i] = b[i] - m[i] * sc;
    }
}

// ---------- weight f32 OIHW -> bf16 [O][kh][kw][I] ----------
__global__ void wperm_kernel(const float* __restrict__ w, unsigned short* __restrict__ o,
                             int Cout, int Cin, int ks) {
    int n = Cout * Cin * ks * ks;
    int i = blockIdx.x * blockDim.x + threadIdx.x;
    if (i >= n) return;
    int cish = __ffs(Cin) - 1;                 // Cin is a power of two
    int ci = i & (Cin - 1); int q = i >> cish;
    int kw = q % ks;  q /= ks;                 // ks = 1 or 3 (tiny kernel, div ok)
    int kh = q % ks;  int co = q / ks;
    o[i] = f2bf(w[((co * Cin + ci) * ks + kh) * ks + kw]);
}

// ---------- fused Haar DWT + 1x1 fusion conv (12->64) + BN + ReLU ----------
// x: (N,3,IH,IW) f32 NCHW ; out: NHWC bf16 (N, IH/2, IW/2, 64)
__global__ void stem_kernel(const float* __restrict__ x, const float* __restrict__ wf,
                            const float* __restrict__ s, const float* __restrict__ t,
                            unsigned short* __restrict__ out, int N, int IH, int IW) {
    int OH = IH >> 1, OW = IW >> 1;             // powers of two
    int owsh = __ffs(OW) - 1, ohsh = __ffs(OH) - 1;
    size_t total = (size_t)N * OH * OW;
    size_t i = (size_t)blockIdx.x * blockDim.x + threadIdx.x;
    if (i >= total) return;
    int ox = (int)(i & (OW - 1)); size_t q = i >> owsh;
    int oy = (int)(q & (OH - 1)); int b = (int)(q >> ohsh);

    float feat[12];
#pragma unroll
    for (int c = 0; c < 3; ++c) {
        const float* px = x + (((size_t)b * 3 + c) * IH + 2 * (size_t)oy) * IW + 2 * (size_t)ox;
        float a = px[0], bb = px[1], cc = px[IW], dd = px[IW + 1];
        feat[c * 4 + 0] = (a + bb + cc + dd) * 0.5f;
        feat[c * 4 + 1] = (a + bb - cc - dd) * 0.5f;
        feat[c * 4 + 2] = (a - bb + cc - dd) * 0.5f;
        feat[c * 4 + 3] = (a - bb - cc + dd) * 0.5f;
    }
    unsigned short* op = out + i * 64;
    for (int co = 0; co < 64; ++co) {
        const float* w = wf + co * 12;
        float acc = 0.f;
#pragma unroll
        for (int k = 0; k < 12; ++k) acc += w[k] * feat[k];
        acc = acc * s[co] + t[co];
        if (acc < 0.f) acc = 0.f;
        op[co] = f2bf(acc);
    }
}

// ---------- maxpool 3x3 stride2 pad1, NHWC bf16 -> NHWC bf16 ----------
__global__ void maxpool_kernel(const unsigned short* __restrict__ in, unsigned short* __restrict__ out,
                               int N, int C, int H, int W) {
    int OH = H >> 1, OW = W >> 1;               // powers of two, as is C
    int csh = __ffs(C) - 1, owsh = __ffs(OW) - 1, ohsh = __ffs(OH) - 1;
    size_t total = (size_t)N * OH * OW * C;
    size_t i = (size_t)blockIdx.x * blockDim.x + threadIdx.x;
    if (i >= total) return;
    int c = (int)(i & (C - 1)); size_t q = i >> csh;
    int ox = (int)(q & (OW - 1)); q >>= owsh;
    int oy = (int)(q & (OH - 1)); int b = (int)(q >> ohsh);
    float m = -3.4e38f;
#pragma unroll
    for (int dy = -1; dy <= 1; ++dy) {
#pragma unroll
        for (int dx = -1; dx <= 1; ++dx) {
            int iy = 2 * oy + dy, ix = 2 * ox + dx;
            if ((unsigned)iy < (unsigned)H && (unsigned)ix < (unsigned)W) {
                float v = bf2f(in[(((size_t)b * H + iy) * W + ix) * C + c]);
                m = v > m ? v : m;
            }
        }
    }
    out[i] = f2bf(m);
}

// ---------- implicit-GEMM conv with WMMA bf16 (NHWC) ----------
// GEMM: M = N*OH*OW, Ndim = Cout, K = ks*ks*Cin ordered (kh,kw,ci).
// Block tile 128(M) x 64(N), 8 waves in 4x2, each wave 2x2 WMMA tiles.
// Staging uses CDNA5 GLOBAL_LOAD_ASYNC_TO_LDS_B128 (ASYNCcnt-tracked DMA,
// no VGPR hop), ping-pong LDS, one barrier per K-step.
#define TM 128
#define TN 64
#define TK 32

__global__ __launch_bounds__(256)
void conv_wmma_kernel(const unsigned short* __restrict__ in, const unsigned short* __restrict__ wbf,
                      const float* __restrict__ scale, const float* __restrict__ bias,
                      const unsigned short* __restrict__ residual,
                      unsigned short* __restrict__ outbf, float* __restrict__ outf,
                      int N, int Cin, int H, int W, int Cout, int OH, int OW,
                      int ks, int pad, int stride, int relu) {
    __shared__ unsigned short As[2][TM][TK];   // ping-pong [m][k]
    __shared__ unsigned short Bs[2][TN][TK];   // ping-pong [n][k]

    const int tid  = threadIdx.x;
    const int lane = tid & 31;
    const int wave = tid >> 5;          // 8 waves
    const int wm   = wave >> 1;         // 0..3 : 32-row M strip
    const int wn   = wave & 1;          // 0..1 : 32-col N strip
    const int m0   = blockIdx.x * TM;
    const int n0   = blockIdx.y * TN;
    const int Ktot = Cin * ks * ks;

    // OW/OH are powers of two for every conv in this network -> shift decode
    const int owsh = __ffs(OW) - 1, ohsh = __ffs(OH) - 1;
    const int owm = OW - 1, ohm = OH - 1;

    // ---- hoisted per-thread staging decode ----
    // A staging: thread handles row ml = tid>>1, 16 contiguous K at kl0 = (tid&1)*16
    const int a_ml  = tid >> 1;
    const int a_kl0 = (tid & 1) * 16;
    const int am_ = m0 + a_ml;
    const int a_ow = am_ & owm;  const int a_t = am_ >> owsh;
    const int a_oh = a_t & ohm;  const int a_b = a_t >> ohsh;
    const int a_oys = a_oh * stride - pad;
    const int a_oxs = a_ow * stride - pad;
    const unsigned a_rowbase = (unsigned)a_b * (unsigned)H * (unsigned)W;

    // B staging: thread handles row nl = tid>>2, 8 contiguous K at bkl0 = (tid&3)*8
    const int b_nl  = tid >> 2;
    const int b_kl0 = (tid & 3) * 8;
    const unsigned b_voff0 = (unsigned)(((size_t)(n0 + b_nl) * Ktot + b_kl0) * 2);

    // uniform 64-bit bases for GVS-mode async loads (SADDR scalar + 32-bit VADDR)
    const unsigned long long in_base = (unsigned long long)in;
    const unsigned long long w_base  = (unsigned long long)wbf;

    // per-thread LDS byte addresses (low 32 bits of generic address = LDS offset)
    const unsigned a_lds0 = (unsigned)(unsigned long long)&As[0][a_ml][a_kl0];
    const unsigned a_lds1 = (unsigned)(unsigned long long)&As[1][a_ml][a_kl0];
    const unsigned b_lds0 = (unsigned)(unsigned long long)&Bs[0][b_nl][b_kl0];
    const unsigned b_lds1 = (unsigned)(unsigned long long)&Bs[1][b_nl][b_kl0];

    auto issueA = [&](int kh, int kw, int ci0, int dstbuf) {
        int iy = a_oys + kh;
        int ix = a_oxs + kw;
        unsigned lds = dstbuf ? a_lds1 : a_lds0;
        if ((unsigned)iy < (unsigned)H && (unsigned)ix < (unsigned)W) {
            unsigned voff = (a_rowbase + (unsigned)iy * (unsigned)W + (unsigned)ix)
                                * (unsigned)Cin + (unsigned)(ci0 + a_kl0);
            voff *= 2u;   // bytes
            asm volatile("global_load_async_to_lds_b128 %0, %1, %2"
                         :: "v"(lds), "v"(voff), "s"(in_base) : "memory");
            asm volatile("global_load_async_to_lds_b128 %0, %1, %2"
                         :: "v"(lds + 16u), "v"(voff + 16u), "s"(in_base) : "memory");
        } else {
            uint4 z = make_uint4(0, 0, 0, 0);
            uint4* sp = (uint4*)&As[dstbuf][a_ml][a_kl0];
            sp[0] = z; sp[1] = z;
        }
    };
    auto issueB = [&](int k0, int dstbuf) {
        unsigned lds  = dstbuf ? b_lds1 : b_lds0;
        unsigned voff = b_voff0 + (unsigned)k0 * 2u;
        asm volatile("global_load_async_to_lds_b128 %0, %1, %2"
                     :: "v"(lds), "v"(voff), "s"(w_base) : "memory");
    };

    v8f acc[2][2] = {};

    // ---- prologue: DMA K-step 0 into buffer 0 ----
    int kh = 0, kw = 0, ci0 = 0;
    issueA(kh, kw, ci0, 0);
    issueB(0, 0);
    asm volatile("s_wait_asynccnt 0x0" ::: "memory");
    __syncthreads();

    int buf = 0;
    for (int k0 = 0; k0 < Ktot; k0 += TK) {
        // next-slab indices (K ordered as kh, kw, ci)
        int nci0 = ci0 + TK, nkw = kw, nkh = kh;
        if (nci0 == Cin) { nci0 = 0; if (++nkw == ks) { nkw = 0; ++nkh; } }
        const bool more = (k0 + TK) < Ktot;

        // ---- kick off DMA for next slab into alternate buffer ----
        if (more) {
            issueA(nkh, nkw, nci0, buf ^ 1);
            issueB(k0 + TK, buf ^ 1);
            if (k0 + 2 * TK < Ktot)
                __builtin_prefetch(wbf + ((size_t)(n0 + b_nl) * Ktot + b_kl0 + k0 + 2 * TK), 0, 1);
        }

        // ---- fragments in contiguous k order (k = 16*(lane/16) + s) for A and B + 2x2 WMMA (overlaps DMA) ----
        union AF { v16bf v; v8bf h[2]; } af[2];
        v16bf bf_[2];
        const int ab = (lane >> 4) * 16;    // A: lanes 0-15 K0-15, 16-31 K16-31
        const int bb = (lane >> 4) * 16;    // B: lanes 0-15 K0-15, 16-31 K16-31
#pragma unroll
        for (int sm = 0; sm < 2; ++sm) {
            int am = wm * 32 + sm * 16 + (lane & 15);
            af[sm].h[0] = *(const v8bf*)&As[buf][am][ab];
            af[sm].h[1] = *(const v8bf*)&As[buf][am][ab + 8];
        }
#pragma unroll
        for (int sn = 0; sn < 2; ++sn) {
            int bn_ = wn * 32 + sn * 16 + (lane & 15);
            bf_[sn] = *(const v16bf*)&Bs[buf][bn_][bb];
        }
#pragma unroll
        for (int sm = 0; sm < 2; ++sm)
#pragma unroll
            for (int sn = 0; sn < 2; ++sn)
                acc[sm][sn] = __builtin_amdgcn_wmma_f32_16x16x32_bf16(
                    false, af[sm].v, false, bf_[sn], (short)0, acc[sm][sn], false, false);

        // ---- complete own DMA, then block-wide barrier (ping-pong) ----
        asm volatile("s_wait_asynccnt 0x0" ::: "memory");
        __syncthreads();

        buf ^= 1; ci0 = nci0; kw = nkw; kh = nkh;
    }

    // ---- epilogue: BN fold + optional residual + ReLU (NHWC out, optional NCHW f32) ----
#pragma unroll
    for (int sn = 0; sn < 2; ++sn) {
        int n = n0 + wn * 32 + sn * 16 + (lane & 15);
        float sc = scale[n], bi = bias[n];
#pragma unroll
        for (int sm = 0; sm < 2; ++sm) {
#pragma unroll
            for (int r = 0; r < 8; ++r) {
                int m  = m0 + wm * 32 + sm * 16 + r + ((lane >> 4) * 8);
                int ow = m & owm;  int tt = m >> owsh;
                int oh = tt & ohm; int b  = tt >> ohsh;
                float v = acc[sm][sn][r] * sc + bi;
                size_t oidx = (((size_t)b * OH + oh) * OW + ow) * Cout + n;   // NHWC
                if (residual) v += bf2f(residual[oidx]);
                if (relu && v < 0.f) v = 0.f;
                outbf[oidx] = f2bf(v);
                if (outf) outf[(((size_t)b * Cout + n) * OH + oh) * OW + ow] = v;   // NCHW
            }
        }
    }
}

// ---------- host orchestration ----------
extern "C" void kernel_launch(void* const* d_in, const int* in_sizes, int n_in,
                              void* d_out, int out_size, void* d_ws, size_t ws_size,
                              hipStream_t stream) {
    (void)in_sizes; (void)n_in; (void)out_size; (void)ws_size;

    // --- input unpack (setup_inputs dict order) ---
    const float* x        = (const float*)d_in[0];
    const float* fusion_w = (const float*)d_in[1];
    const float* stem_bn[4];
    int p = 2;
    for (int i = 0; i < 4; ++i) stem_bn[i] = (const float*)d_in[p++];

    struct BlockP {
        const float *c1w, *c2w, *dsw;
        const float *bn1[4], *bn2[4], *dsbn[4];
        int cin, cout, stride, hasds;
    };
    const int cfg[4][4] = {{64,64,1,2},{64,128,2,2},{128,256,2,2},{256,512,2,2}};
    BlockP blocks[8]; int nb = 0;
    for (int li = 0; li < 4; ++li) {
        int cin0 = cfg[li][0], cout = cfg[li][1], stride = cfg[li][2], nblk = cfg[li][3];
        for (int bi = 0; bi < nblk; ++bi) {
            BlockP& B = blocks[nb++];
            B.cin = bi ? cout : cin0;
            B.cout = cout;
            B.stride = bi ? 1 : stride;
            B.hasds = (bi == 0 && (B.stride != 1 || B.cin != cout)) ? 1 : 0;
            B.c1w = (const float*)d_in[p++];
            for (int i = 0; i < 4; ++i) B.bn1[i] = (const float*)d_in[p++];
            B.c2w = (const float*)d_in[p++];
            for (int i = 0; i < 4; ++i) B.bn2[i] = (const float*)d_in[p++];
            if (B.hasds) {
                B.dsw = (const float*)d_in[p++];
                for (int i = 0; i < 4; ++i) B.dsbn[i] = (const float*)d_in[p++];
            } else {
                B.dsw = nullptr;
            }
        }
    }

    // --- workspace carve-out (identical every call -> deterministic) ---
    char* wsb = (char*)d_ws;
    size_t off = 0;
    auto alloc = [&](size_t bytes) -> void* {
        void* r = wsb + off;
        off += (bytes + 255) & ~(size_t)255;
        return r;
    };

    const int NBATCH = 16;
    unsigned short* stem_act = (unsigned short*)alloc((size_t)NBATCH * 64 * 256 * 256 * 2);
    unsigned short* bufA = (unsigned short*)alloc((size_t)NBATCH * 64 * 128 * 128 * 2);
    unsigned short* bufB = (unsigned short*)alloc((size_t)NBATCH * 64 * 128 * 128 * 2);
    unsigned short* bufC = (unsigned short*)alloc((size_t)NBATCH * 64 * 128 * 128 * 2);
    unsigned short* idn  = (unsigned short*)alloc((size_t)NBATCH * 128 * 64 * 64 * 2);

    auto wprep = [&](const float* w, int Cout, int Cin, int ks) -> unsigned short* {
        int n = Cout * Cin * ks * ks;
        unsigned short* o = (unsigned short*)alloc((size_t)n * 2);
        wperm_kernel<<<(n + 255) / 256, 256, 0, stream>>>(w, o, Cout, Cin, ks);
        return o;
    };
    auto bnfold = [&](const float* const bn[4], int c, float*& s, float*& t) {
        s = (float*)alloc((size_t)c * 4);
        t = (float*)alloc((size_t)c * 4);
        bnfold_kernel<<<(c + 255) / 256, 256, 0, stream>>>(bn[0], bn[1], bn[2], bn[3], s, t, c);
    };
    auto conv = [&](const unsigned short* in, const unsigned short* wbf,
                    const float* s, const float* t, const unsigned short* res,
                    unsigned short* outb, float* outf,
                    int Cin, int H, int Cout, int ks, int pad, int stride, int relu) {
        int OH = (H + 2 * pad - ks) / stride + 1;
        int M = NBATCH * OH * OH;
        dim3 g((unsigned)(M / TM), (unsigned)(Cout / TN));
        conv_wmma_kernel<<<g, 256, 0, stream>>>(in, wbf, s, t, res, outb, outf,
                                                NBATCH, Cin, H, H, Cout, OH, OH,
                                                ks, pad, stride, relu);
    };

    // --- stem: DWT + fusion conv + BN + ReLU, then maxpool (all NHWC) ---
    float *ss, *st;
    bnfold(stem_bn, 64, ss, st);
    {
        size_t tot = (size_t)NBATCH * 256 * 256;   // one thread per pixel
        stem_kernel<<<(unsigned)((tot + 255) / 256), 256, 0, stream>>>(
            x, fusion_w, ss, st, stem_act, NBATCH, 512, 512);
        size_t ptot = (size_t)NBATCH * 128 * 128 * 64;
        maxpool_kernel<<<(unsigned)((ptot + 255) / 256), 256, 0, stream>>>(
            stem_act, bufA, NBATCH, 64, 256, 256);
    }

    // --- residual layers ---
    float* outp = (float*)d_out;
    const size_t outoff[4] = {0, 16777216, 25165824, 29360128};

    unsigned short* bin  = bufA;
    unsigned short* btmp = bufB;
    unsigned short* bout = bufC;
    int H = 128;
    int blk = 0;
    for (int li = 0; li < 4; ++li) {
        int nblk = cfg[li][3];
        for (int bi = 0; bi < nblk; ++bi) {
            BlockP& B = blocks[blk++];
            int OH = H / B.stride;

            unsigned short* w1 = wprep(B.c1w, B.cout, B.cin, 3);
            float *s1, *t1; bnfold(B.bn1, B.cout, s1, t1);
            conv(bin, w1, s1, t1, nullptr, btmp, nullptr,
                 B.cin, H, B.cout, 3, 1, B.stride, 1);

            const unsigned short* res;
            if (B.hasds) {
                unsigned short* wd = wprep(B.dsw, B.cout, B.cin, 1);
                float *sd, *td; bnfold(B.dsbn, B.cout, sd, td);
                conv(bin, wd, sd, td, nullptr, idn, nullptr,
                     B.cin, H, B.cout, 1, 0, B.stride, 0);
                res = idn;
            } else {
                res = bin;
            }

            unsigned short* w2 = wprep(B.c2w, B.cout, B.cout, 3);
            float *s2, *t2; bnfold(B.bn2, B.cout, s2, t2);
            float* of = (bi == nblk - 1) ? (outp + outoff[li]) : nullptr;
            conv(btmp, w2, s2, t2, res, bout, of,
                 B.cout, OH, B.cout, 3, 1, 1, 1);

            // rotate buffers: output becomes next input
            unsigned short* t = bin; bin = bout; bout = t;
            H = OH;
        }
    }
}
